// GraphTransformerLayer_72524817760577
// MI455X (gfx1250) — matrix operand order repaired and checked by reading, not compiled, not run on hardware
//
#include <hip/hip_runtime.h>
#include <hip/hip_bf16.h>
#include <stdint.h>

// ---------------------------------------------------------------------------
// Graph transformer layer for MI455X (gfx1250), wave32 + WMMA bf16.
//   N=4096 tokens, D_MODEL=512, 8 heads x d_k=64, D_FF=2048.
// Pipeline:
//   adj -> packed bitmask (ballot, read adj from HBM exactly once)
//   f32->bf16 casts -> QKV WMMA GEMMs (Q pre-scaled by 1/8, V transposed) ->
//   flash attention (64-key blocks, online softmax, bitmask, LDS P-transpose)
//   -> Wo GEMM -> residual+LN1 -> FFN GEMM(relu) -> FFN GEMM -> residual+LN2.
// ---------------------------------------------------------------------------

typedef __attribute__((ext_vector_type(16))) __bf16 v16bf;
typedef __attribute__((ext_vector_type(8)))  __bf16 v8bf;
typedef __attribute__((ext_vector_type(8)))  float  v8f;

#define NTOK   4096
#define DMODEL 512
#define NHEADS 8
#define DKH    64
#define DFF    2048
#define NWORDS (NTOK / 32)

__device__ __forceinline__ v8f wmma_bf16(v16bf a, v16bf b, v8f c) {
  // D = A(16x32 bf16) * B(32x16 bf16) + C(16x16 f32)
  return __builtin_amdgcn_wmma_f32_16x16x32_bf16(
      /*neg_a=*/false, a, /*neg_b=*/false, b,
      /*c_mod=*/(short)0, c, /*reuse_a=*/false, /*reuse_b=*/false);
}

// A-fragment (16x32, 16-bit): lane l -> row l%16; lanes 0-15 hold K 0..15,
// lanes 16-31 hold K 16..31 (contiguous per lane).  p points at
// row_base + (lane>>4)*16; elems 8..15 come from +8 elements.
__device__ __forceinline__ v16bf ld_afrag(const __bf16* p) {
  union { v16bf v; v8bf h[2]; } u;
  u.h[0] = *(const v8bf*)(p);
  u.h[1] = *(const v8bf*)(p + 8);
  return u.v;
}
// B-fragment (32x16, 16-bit): lane l -> col l%16; lanes 0-15 hold K 0..15,
// lanes 16-31 hold K 16..31 (contiguous per lane).
__device__ __forceinline__ v16bf ld_bfrag(const __bf16* p) {
  union { v16bf v; v8bf h[2]; } u;
  u.h[0] = *(const v8bf*)(p);
  u.h[1] = *(const v8bf*)(p + 8);
  return u.v;
}

// ---------------------------------------------------------------------------
__global__ void cast_bf16_kernel(const float* __restrict__ s,
                                 __bf16* __restrict__ d, int n) {
  int i = blockIdx.x * blockDim.x + threadIdx.x;
  int stride = gridDim.x * blockDim.x;
  for (; i < n; i += stride) d[i] = (__bf16)s[i];
}

// ---------------------------------------------------------------------------
// Pack adj (int32, 0/1) into a bitmask: word[q][w] bit j = (adj[q][32w+j]!=0).
// One wave packs 32 consecutive elements of a row via ballot.
__global__ void pack_mask_kernel(const int* __restrict__ adj,
                                 unsigned* __restrict__ m) {
  const int i = blockIdx.x * blockDim.x + threadIdx.x;   // element index
  const bool pred = adj[i] != 0;
#if defined(__has_builtin) && __has_builtin(__builtin_amdgcn_ballot_w32)
  const unsigned word = __builtin_amdgcn_ballot_w32(pred);
#else
  const unsigned word = (unsigned)__ballot(pred);
#endif
  if ((threadIdx.x & 31) == 0) m[i >> 5] = word;
}

// ---------------------------------------------------------------------------
// C[M,N] = act((A[M,K] @ W[N,K]^T + bias) * oscale). One wave -> 16x64 of C
// (4 WMMA accumulators); block = 4 waves -> 16x256. OM: 0=bf16, 1=bf16
// transposed (C^T stored as [N][M], 16B contiguous per lane), 2=f32.
template<int OM, bool RELU>
__global__ void __launch_bounds__(128) gemm_xwt_kernel(
    const __bf16* __restrict__ A, const __bf16* __restrict__ W,
    const float* __restrict__ bias, void* __restrict__ Cp,
    int M, int N, int K, float oscale)
{
  const int w    = threadIdx.x >> 5;
  const int lane = threadIdx.x & 31;
  const int r    = lane & 15;    // A row within tile / C col within 16-chunk
  const int hi   = lane >> 4;    // lane half
  const int row0 = blockIdx.x * 16;
  const int col0 = (blockIdx.y * 4 + w) * 64;

  v8f acc[4] = {};
  const __bf16* ap = A + (size_t)(row0 + r) * K + hi * 16;

  for (int kk = 0; kk < K; kk += 32) {
    v16bf af = ld_afrag(ap + kk);
#pragma unroll
    for (int t = 0; t < 4; ++t) {
      const __bf16* bp = W + (size_t)(col0 + t * 16 + r) * K + kk + hi * 16;
      acc[t] = wmma_bf16(af, ld_bfrag(bp), acc[t]);
    }
  }

#pragma unroll
  for (int t = 0; t < 4; ++t) {
    float bz = bias[col0 + t * 16 + r];
#pragma unroll
    for (int e = 0; e < 8; ++e) {
      float v = (acc[t][e] + bz) * oscale;
      if (RELU) v = fmaxf(v, 0.0f);
      acc[t][e] = v;
    }
  }

  if constexpr (OM == 0) {                       // bf16 row-major
    __bf16* C = (__bf16*)Cp;
#pragma unroll
    for (int t = 0; t < 4; ++t)
#pragma unroll
      for (int e = 0; e < 8; ++e)
        C[(size_t)(row0 + e + 8 * hi) * N + col0 + t * 16 + r] =
            (__bf16)acc[t][e];
  } else if constexpr (OM == 1) {                // bf16 transposed: C[N][M]
    __bf16* C = (__bf16*)Cp;
#pragma unroll
    for (int t = 0; t < 4; ++t) {
      v8bf o;
#pragma unroll
      for (int e = 0; e < 8; ++e) o[e] = (__bf16)acc[t][e];
      *(v8bf*)(C + (size_t)(col0 + t * 16 + r) * M + row0 + 8 * hi) = o;
    }
  } else {                                       // f32 row-major
    float* C = (float*)Cp;
#pragma unroll
    for (int t = 0; t < 4; ++t)
#pragma unroll
      for (int e = 0; e < 8; ++e)
        C[(size_t)(row0 + e + 8 * hi) * N + col0 + t * 16 + r] = acc[t][e];
  }
}

// ---------------------------------------------------------------------------
// Flash attention with packed adjacency bitmask. Block = 256 threads = 8
// waves; wave w handles head w for query rows [blockIdx.x*16, +16).
// 64-key blocks: 4 score accumulators, P staged as two A-fragments; one
// broadcast b64 mask load per row per block; stat updates amortized 2x.
// Q is pre-scaled by 1/sqrt(d_k) at projection time.
__global__ void __launch_bounds__(256) attn_kernel(
    const __bf16* __restrict__ Q, const __bf16* __restrict__ Km,
    const __bf16* __restrict__ Vt,  // [DMODEL][NTOK] (transposed V, bf16)
    const unsigned* __restrict__ maskbits,  // [NTOK][NWORDS]
    __bf16* __restrict__ ctx)
{
  __shared__ __align__(16) __bf16 pbuf[NHEADS * 16 * 64];   // 16 KiB

  const int w    = threadIdx.x >> 5;   // head
  const int lane = threadIdx.x & 31;
  const int r    = lane & 15;
  const int hi   = lane >> 4;
  const int q0   = blockIdx.x * 16;
  const int hcol = w * DKH;
  __bf16* pb = pbuf + w * (16 * 64);

  // Q A-fragments for d 0..31 and 32..63 of this head (live all loop long)
  const __bf16* qrow = Q + (size_t)(q0 + r) * DMODEL + hcol + hi * 16;
  v16bf qa0 = ld_afrag(qrow);
  v16bf qa1 = ld_afrag(qrow + 32);

  v8f o[4] = {};
  float mrow[8], lrow[8];
#pragma unroll
  for (int e = 0; e < 8; ++e) { mrow[e] = -1e30f; lrow[e] = 0.0f; }

  for (int kb = 0; kb < NTOK; kb += 64) {
    // ---- S = Q K^T for 64 keys (four 16x16 accumulators)
    v8f s[4];
#pragma unroll
    for (int c = 0; c < 4; ++c) {
      const __bf16* kp = Km + (size_t)(kb + c * 16 + r) * DMODEL + hcol + hi * 16;
      v8f t = {};
      t = wmma_bf16(qa0, ld_bfrag(kp), t);
      t = wmma_bf16(qa1, ld_bfrag(kp + 32), t);
      s[c] = t;
    }

    // ---- mask (bitmask + self-loop), online softmax over 64 keys
#pragma unroll
    for (int e = 0; e < 8; ++e) {
      const int q = q0 + e + 8 * hi;
      // 64 mask bits for this row/key-block in one broadcast b64 load
      const uint2 mw = *(const uint2*)(maskbits + (size_t)q * NWORDS + (kb >> 5));
      float v[4];
      {
        const bool k0 = ((mw.x >> r) & 1u)        || (q == kb + r);
        const bool k1 = ((mw.x >> (16 + r)) & 1u) || (q == kb + 16 + r);
        const bool k2 = ((mw.y >> r) & 1u)        || (q == kb + 32 + r);
        const bool k3 = ((mw.y >> (16 + r)) & 1u) || (q == kb + 48 + r);
        v[0] = k0 ? s[0][e] : -1e9f;
        v[1] = k1 ? s[1][e] : -1e9f;
        v[2] = k2 ? s[2][e] : -1e9f;
        v[3] = k3 ? s[3][e] : -1e9f;
      }

      float mx = fmaxf(fmaxf(v[0], v[1]), fmaxf(v[2], v[3]));
      mx = fmaxf(mx, __shfl_xor(mx, 1, 32));
      mx = fmaxf(mx, __shfl_xor(mx, 2, 32));
      mx = fmaxf(mx, __shfl_xor(mx, 4, 32));
      mx = fmaxf(mx, __shfl_xor(mx, 8, 32));

      const float mn = fmaxf(mrow[e], mx);
      const float al = __expf(mrow[e] - mn);
      float pv[4];
      pv[0] = __expf(v[0] - mn);
      pv[1] = __expf(v[1] - mn);
      pv[2] = __expf(v[2] - mn);
      pv[3] = __expf(v[3] - mn);
      float rs = (pv[0] + pv[1]) + (pv[2] + pv[3]);
      rs += __shfl_xor(rs, 1, 32);
      rs += __shfl_xor(rs, 2, 32);
      rs += __shfl_xor(rs, 4, 32);
      rs += __shfl_xor(rs, 8, 32);

      lrow[e] = lrow[e] * al + rs;
      mrow[e] = mn;
      o[0][e] *= al; o[1][e] *= al; o[2][e] *= al; o[3][e] *= al;

      // stage P (C-layout -> LDS row-major 16x64 tile, bf16)
      __bf16* prow = pb + (e + 8 * hi) * 64;
#pragma unroll
      for (int c = 0; c < 4; ++c) prow[c * 16 + r] = (__bf16)pv[c];
    }

    asm volatile("s_wait_dscnt 0" ::: "memory");   // wave-local LDS RAW

    // ---- read P back as two A-fragments, accumulate O += P @ V
    v16bf pf0 = ld_afrag(pb + r * 64 + hi * 16);
    v16bf pf1 = ld_afrag(pb + r * 64 + 32 + hi * 16);
#pragma unroll
    for (int t = 0; t < 4; ++t) {
      const __bf16* vp = Vt + (size_t)(hcol + t * 16 + r) * NTOK + kb + hi * 16;
      o[t] = wmma_bf16(pf0, ld_bfrag(vp), o[t]);
      o[t] = wmma_bf16(pf1, ld_bfrag(vp + 32), o[t]);
    }
  }

  // ---- finalize: divide by row sums, write ctx (bf16 [NTOK][DMODEL])
#pragma unroll
  for (int e = 0; e < 8; ++e) {
    const float inv = 1.0f / lrow[e];
    const size_t row = (size_t)(q0 + e + 8 * hi);
#pragma unroll
    for (int t = 0; t < 4; ++t)
      ctx[row * DMODEL + hcol + t * 16 + r] = (__bf16)(o[t][e] * inv);
  }
}

// ---------------------------------------------------------------------------
// out = LayerNorm(x + res) * g + b; also emits bf16 copy if outb != null.
// One block per row (D=512), 256 threads, 2 elements each.
__global__ void __launch_bounds__(256) resid_ln_kernel(
    const float* __restrict__ x, const float* __restrict__ res,
    const float* __restrict__ g, const float* __restrict__ b,
    float* __restrict__ out, __bf16* __restrict__ outb)
{
  __shared__ float sd[256];
  const int row = blockIdx.x, tid = threadIdx.x;
  const size_t base = (size_t)row * DMODEL;

  float v0 = x[base + tid]       + res[base + tid];
  float v1 = x[base + tid + 256] + res[base + tid + 256];

  sd[tid] = v0 + v1;
  __syncthreads();
  for (int s = 128; s > 0; s >>= 1) {
    if (tid < s) sd[tid] += sd[tid + s];
    __syncthreads();
  }
  const float mean = sd[0] * (1.0f / DMODEL);
  __syncthreads();

  const float d0 = v0 - mean, d1 = v1 - mean;
  sd[tid] = d0 * d0 + d1 * d1;
  __syncthreads();
  for (int s = 128; s > 0; s >>= 1) {
    if (tid < s) sd[tid] += sd[tid + s];
    __syncthreads();
  }
  const float rstd = rsqrtf(sd[0] * (1.0f / DMODEL) + 1e-5f);

  const float o0 = d0 * rstd * g[tid]       + b[tid];
  const float o1 = d1 * rstd * g[tid + 256] + b[tid + 256];
  out[base + tid]       = o0;
  out[base + tid + 256] = o1;
  if (outb) {
    outb[base + tid]       = (__bf16)o0;
    outb[base + tid + 256] = (__bf16)o1;
  }
}

// ---------------------------------------------------------------------------
extern "C" void kernel_launch(void* const* d_in, const int* in_sizes, int n_in,
                              void* d_out, int out_size, void* d_ws,
                              size_t ws_size, hipStream_t stream)
{
  (void)in_sizes; (void)n_in; (void)out_size; (void)ws_size;

  const float* h   = (const float*)d_in[0];
  const int*   adj = (const int*)  d_in[1];
  const float* Wq  = (const float*)d_in[2];
  const float* bq  = (const float*)d_in[3];
  const float* Wk  = (const float*)d_in[4];
  const float* bk  = (const float*)d_in[5];
  const float* Wv  = (const float*)d_in[6];
  const float* bv  = (const float*)d_in[7];
  const float* Wo  = (const float*)d_in[8];
  const float* bo  = (const float*)d_in[9];
  const float* W1  = (const float*)d_in[10];
  const float* b1  = (const float*)d_in[11];
  const float* W2  = (const float*)d_in[12];
  const float* b2  = (const float*)d_in[13];
  const float* g1  = (const float*)d_in[14];
  const float* be1 = (const float*)d_in[15];
  const float* g2  = (const float*)d_in[16];
  const float* be2 = (const float*)d_in[17];
  float* out = (float*)d_out;

  // ---- workspace carve (256B aligned slices)
  char* p = (char*)d_ws;
  auto take = [&](size_t bytes) -> void* {
    void* r = (void*)p;
    p += (bytes + 255) & ~(size_t)255;
    return r;
  };
  __bf16*   hbf  = (__bf16*)  take((size_t)NTOK * DMODEL * 2);
  __bf16*   wqb  = (__bf16*)  take((size_t)DMODEL * DMODEL * 2);
  __bf16*   wkb  = (__bf16*)  take((size_t)DMODEL * DMODEL * 2);
  __bf16*   wvb  = (__bf16*)  take((size_t)DMODEL * DMODEL * 2);
  __bf16*   wob  = (__bf16*)  take((size_t)DMODEL * DMODEL * 2);
  __bf16*   w1b  = (__bf16*)  take((size_t)DFF * DMODEL * 2);
  __bf16*   w2b  = (__bf16*)  take((size_t)DMODEL * DFF * 2);
  __bf16*   Qb   = (__bf16*)  take((size_t)NTOK * DMODEL * 2);
  __bf16*   Kb   = (__bf16*)  take((size_t)NTOK * DMODEL * 2);
  __bf16*   Vtb  = (__bf16*)  take((size_t)DMODEL * NTOK * 2);  // transposed
  __bf16*   ctxb = (__bf16*)  take((size_t)NTOK * DMODEL * 2);
  float*    attn = (float*)   take((size_t)NTOK * DMODEL * 4);
  float*    h1   = (float*)   take((size_t)NTOK * DMODEL * 4);
  __bf16*   h1b  = (__bf16*)  take((size_t)NTOK * DMODEL * 2);
  __bf16*   ff1  = (__bf16*)  take((size_t)NTOK * DFF * 2);
  float*    ff2  = (float*)   take((size_t)NTOK * DMODEL * 4);
  unsigned* mbit = (unsigned*)take((size_t)NTOK * NWORDS * 4); // 2 MiB

  // ---- adj -> packed bitmask (single HBM pass over 64 MB)
  pack_mask_kernel<<<(NTOK * NTOK) / 256, 256, 0, stream>>>(adj, mbit);

  // ---- f32 -> bf16 casts (everything then lives in L2)
  cast_bf16_kernel<<<2048, 256, 0, stream>>>(h,  hbf, NTOK * DMODEL);
  cast_bf16_kernel<<<512,  256, 0, stream>>>(Wq, wqb, DMODEL * DMODEL);
  cast_bf16_kernel<<<512,  256, 0, stream>>>(Wk, wkb, DMODEL * DMODEL);
  cast_bf16_kernel<<<512,  256, 0, stream>>>(Wv, wvb, DMODEL * DMODEL);
  cast_bf16_kernel<<<512,  256, 0, stream>>>(Wo, wob, DMODEL * DMODEL);
  cast_bf16_kernel<<<2048, 256, 0, stream>>>(W1, w1b, DFF * DMODEL);
  cast_bf16_kernel<<<2048, 256, 0, stream>>>(W2, w2b, DMODEL * DFF);

  const dim3 blk(128);
  // ---- QKV projections (Q pre-scaled by 1/sqrt(d_k); V written transposed)
  gemm_xwt_kernel<0, false><<<dim3(NTOK / 16, DMODEL / 256), blk, 0, stream>>>(
      hbf, wqb, bq, Qb, NTOK, DMODEL, DMODEL, 0.125f);
  gemm_xwt_kernel<0, false><<<dim3(NTOK / 16, DMODEL / 256), blk, 0, stream>>>(
      hbf, wkb, bk, Kb, NTOK, DMODEL, DMODEL, 1.0f);
  gemm_xwt_kernel<1, false><<<dim3(NTOK / 16, DMODEL / 256), blk, 0, stream>>>(
      hbf, wvb, bv, Vtb, NTOK, DMODEL, DMODEL, 1.0f);

  // ---- masked flash attention (8 heads per workgroup, bitmask broadcast)
  attn_kernel<<<NTOK / 16, 256, 0, stream>>>(Qb, Kb, Vtb, mbit, ctxb);

  // ---- output projection, residual + LN1
  gemm_xwt_kernel<2, false><<<dim3(NTOK / 16, DMODEL / 256), blk, 0, stream>>>(
      ctxb, wob, bo, attn, NTOK, DMODEL, DMODEL, 1.0f);
  resid_ln_kernel<<<NTOK, 256, 0, stream>>>(h, attn, g1, be1, h1, h1b);

  // ---- FFN: relu(h1 @ W1^T + b1) @ W2^T + b2, residual + LN2
  gemm_xwt_kernel<0, true><<<dim3(NTOK / 16, DFF / 256), blk, 0, stream>>>(
      h1b, w1b, b1, ff1, NTOK, DFF, DMODEL, 1.0f);
  gemm_xwt_kernel<2, false><<<dim3(NTOK / 16, DMODEL / 256), blk, 0, stream>>>(
      ff1, w2b, b2, ff2, NTOK, DMODEL, DFF, 1.0f);
  resid_ln_kernel<<<NTOK, 256, 0, stream>>>(h1, ff2, g2, be2, out, nullptr);
}
